// EuclideanDeconf_10290741641835
// MI455X (gfx1250) — matrix operand order repaired and checked by reading, not compiled, not run on hardware
//
#include <hip/hip_runtime.h>
#include <hip/hip_bf16.h>
#include <stdint.h>

#define B_DIM 4096
#define D_DIM 4096
#define C_DIM 1024

#define BM 128
#define BN 128
#define BK 32
#define LDK 40   // padded LDS row stride in bf16 elements (80 B, 16-B aligned)

typedef __attribute__((ext_vector_type(16))) __bf16 bf16x16;
typedef __attribute__((ext_vector_type(4)))  __bf16 bf16x4;
typedef __attribute__((ext_vector_type(4)))  float  f32x4;
typedef __attribute__((ext_vector_type(8)))  float  f32x8;
typedef __attribute__((ext_vector_type(4)))  int    i32x4;

#if __has_builtin(__builtin_amdgcn_global_load_async_to_lds_b128)
#define HAVE_ASYNC_LDS 1
typedef i32x4 __attribute__((address_space(1))) g_i32x4;   // global (AS1)
typedef i32x4 __attribute__((address_space(3))) l_i32x4;   // LDS    (AS3)
#endif

__device__ __forceinline__ void wait_async_zero() {
#if __has_builtin(__builtin_amdgcn_s_wait_asynccnt)
    __builtin_amdgcn_s_wait_asynccnt(0);
#else
    asm volatile("s_wait_asynccnt 0" ::: "memory");
#endif
}

// Convert 4 fp32 -> 4 bf16 (RNE) via the native path (v_cvt_pk_bf16_f32).
__device__ __forceinline__ uint2 cvt4_bf16(float4 v) {
    f32x4 f = {v.x, v.y, v.z, v.w};
    union { bf16x4 h; uint2 u; } r;
    r.h = __builtin_convertvector(f, bf16x4);
    return r.u;
}

// ---------------------------------------------------------------------------
// Kernel 1a: squared row norms only (fallback path, tiny workspace).
// ---------------------------------------------------------------------------
__global__ __launch_bounds__(256) void norms_kernel(
    const float* __restrict__ x, const float* __restrict__ W,
    float* __restrict__ xs, float* __restrict__ ws)
{
    const int row = blockIdx.x;
    const float* src;
    float* dst;
    int r;
    if (row < B_DIM) { src = x + (size_t)row * D_DIM;          dst = xs; r = row; }
    else             { src = W + (size_t)(row - B_DIM) * D_DIM; dst = ws; r = row - B_DIM; }

    float s = 0.f;
    for (int c = threadIdx.x * 4; c < D_DIM; c += blockDim.x * 4) {
        float4 v = *(const float4*)(src + c);
        s += v.x * v.x + v.y * v.y + v.z * v.z + v.w * v.w;
    }
    __shared__ float red[256];
    red[threadIdx.x] = s;
    __syncthreads();
    for (int off = 128; off > 0; off >>= 1) {
        if (threadIdx.x < off) red[threadIdx.x] += red[threadIdx.x + off];
        __syncthreads();
    }
    if (threadIdx.x == 0) dst[r] = red[0];
}

// ---------------------------------------------------------------------------
// Kernel 1b: squared row norms + bf16 copies of x and W (async-GEMM path).
// One pass over the fp32 data; afterwards both bf16 operands (40 MB) live in L2.
// ---------------------------------------------------------------------------
__global__ __launch_bounds__(256) void prep_kernel(
    const float* __restrict__ x, const float* __restrict__ W,
    float* __restrict__ xs, float* __restrict__ ws,
    uint16_t* __restrict__ xbf, uint16_t* __restrict__ wbf)
{
    const int row = blockIdx.x;
    const float* src;
    uint16_t* dbf;
    float* dst;
    int r;
    if (row < B_DIM) { src = x + (size_t)row * D_DIM;           dst = xs; dbf = xbf + (size_t)row * D_DIM;           r = row; }
    else             { src = W + (size_t)(row - B_DIM) * D_DIM; dst = ws; dbf = wbf + (size_t)(row - B_DIM) * D_DIM; r = row - B_DIM; }

    float s = 0.f;
    for (int c = threadIdx.x * 4; c < D_DIM; c += blockDim.x * 4) {
        float4 v = *(const float4*)(src + c);
        s += v.x * v.x + v.y * v.y + v.z * v.z + v.w * v.w;
        *(uint2*)(dbf + c) = cvt4_bf16(v);
    }
    __shared__ float red[256];
    red[threadIdx.x] = s;
    __syncthreads();
    for (int off = 128; off > 0; off >>= 1) {
        if (threadIdx.x < off) red[threadIdx.x] += red[threadIdx.x + off];
        __syncthreads();
    }
    if (threadIdx.x == 0) dst[r] = red[0];
}

// ---------------------------------------------------------------------------
// Shared WMMA compute + epilogue helpers (same tiling for both GEMM kernels).
// ---------------------------------------------------------------------------
union Frag { uint4 u[2]; bf16x16 v; };

__device__ __forceinline__ void wmma_step(
    const uint16_t* __restrict__ AsBuf, const uint16_t* __restrict__ BsBuf,
    int lane, int waveM, int waveN, f32x8 acc[2][4])
{
    Frag af[2];
    #pragma unroll
    for (int i = 0; i < 2; ++i) {
        // A 16x32 bf16 layout: lanes<16: K 0-7 & 16-23; lanes>=16: K 8-15 & 24-31
        const int row = waveM * 32 + i * 16 + (lane & 15);
        const int cb  = (lane < 16) ? 0 : 8;
        const uint16_t* p = AsBuf + row * LDK + cb;
        af[i].u[0] = *(const uint4*)p;          // ds_load_b128
        af[i].u[1] = *(const uint4*)(p + 16);   // ds_load_b128
    }
    Frag bfm[4];
    #pragma unroll
    for (int j = 0; j < 4; ++j) {
        // B 32x16 bf16 layout: lanes<16: K 0-7 & 16-23; lanes>=16: K 8-15 & 24-31; N = lane&15
        const int row = waveN * 64 + j * 16 + (lane & 15);
        const int cb  = (lane < 16) ? 0 : 8;
        const uint16_t* p = BsBuf + row * LDK + cb;
        bfm[j].u[0] = *(const uint4*)p;         // ds_load_b128
        bfm[j].u[1] = *(const uint4*)(p + 16);  // ds_load_b128
    }
    #pragma unroll
    for (int i = 0; i < 2; ++i)
        #pragma unroll
        for (int j = 0; j < 4; ++j)
            acc[i][j] = __builtin_amdgcn_wmma_f32_16x16x32_bf16(
                false, af[i].v, false, bfm[j].v,
                (short)0, acc[i][j], false, false);
}

__device__ __forceinline__ void dist_epilogue(
    const float* __restrict__ xs, const float* __restrict__ ws,
    float* __restrict__ out, int blockRow, int blockCol,
    int lane, int waveM, int waveN, f32x8 acc[2][4])
{
    const float invD = 1.0f / (float)D_DIM;
    #pragma unroll
    for (int i = 0; i < 2; ++i) {
        // C/D layout: VGPR v -> M = v (lanes<16) or v+8 (lanes>=16); N = lane&15
        const int rowBase = blockRow + waveM * 32 + i * 16 + ((lane < 16) ? 0 : 8);
        #pragma unroll
        for (int j = 0; j < 4; ++j) {
            const int col = blockCol + waveN * 64 + j * 16 + (lane & 15);
            const float wsq = ws[col];
            #pragma unroll
            for (int v = 0; v < 8; ++v) {
                const int row = rowBase + v;
                const float val = -(xs[row] - 2.0f * acc[i][j][v] + wsq) * invD;
                out[(size_t)row * C_DIM + col] = val;
            }
        }
    }
}

// ---------------------------------------------------------------------------
// Kernel 2a: GEMM with fp32->bf16 conversion fused into LDS staging (fallback).
// ---------------------------------------------------------------------------
__global__ __launch_bounds__(256) void dist_gemm_kernel(
    const float* __restrict__ x, const float* __restrict__ W,
    const float* __restrict__ xs, const float* __restrict__ ws,
    float* __restrict__ out)
{
    __shared__ __align__(16) uint16_t As[2][BM * LDK];
    __shared__ __align__(16) uint16_t Bs[2][BN * LDK];

    const int tid   = threadIdx.x;
    const int lane  = tid & 31;
    const int wave  = tid >> 5;
    const int waveM = wave & 3;
    const int waveN = wave >> 2;
    const int blockRow = blockIdx.x * BM;
    const int blockCol = blockIdx.y * BN;

    const int ldRow = tid >> 3;         // 0..31
    const int ldCol = (tid & 7) * 4;    // 0,4,...,28

    f32x8 acc[2][4];
    #pragma unroll
    for (int i = 0; i < 2; ++i)
        #pragma unroll
        for (int j = 0; j < 4; ++j)
            acc[i][j] = (f32x8){};

    auto load_tile = [&](int buf, int kt) {
        const int k0 = kt * BK;
        #pragma unroll
        for (int rr = 0; rr < 4; ++rr) {
            const int r = ldRow + rr * 32;
            float4 va = *(const float4*)(x + (size_t)(blockRow + r) * D_DIM + k0 + ldCol);
            float4 vb = *(const float4*)(W + (size_t)(blockCol + r) * D_DIM + k0 + ldCol);
            *(uint2*)&As[buf][r * LDK + ldCol] = cvt4_bf16(va);
            *(uint2*)&Bs[buf][r * LDK + ldCol] = cvt4_bf16(vb);
        }
    };

    const int NK = D_DIM / BK;
    load_tile(0, 0);
    __syncthreads();
    for (int kt = 0; kt < NK; ++kt) {
        const int buf = kt & 1;
        if (kt + 1 < NK) load_tile(buf ^ 1, kt + 1);
        wmma_step(As[buf], Bs[buf], lane, waveM, waveN, acc);
        __syncthreads();
    }
    dist_epilogue(xs, ws, out, blockRow, blockCol, lane, waveM, waveN, acc);
}

// ---------------------------------------------------------------------------
// Kernel 2b: GEMM streaming pre-converted bf16 tiles via async global->LDS
// (GLOBAL_LOAD_ASYNC_TO_LDS_B128, ASYNCcnt) — zero staging VALU/VGPR traffic.
// ---------------------------------------------------------------------------
__global__ __launch_bounds__(256) void dist_gemm_async_kernel(
    const uint16_t* __restrict__ xbf, const uint16_t* __restrict__ wbf,
    const float* __restrict__ xs, const float* __restrict__ ws,
    float* __restrict__ out)
{
    __shared__ __align__(16) uint16_t As[2][BM * LDK];
    __shared__ __align__(16) uint16_t Bs[2][BN * LDK];

    const int tid   = threadIdx.x;
    const int lane  = tid & 31;
    const int wave  = tid >> 5;
    const int waveM = wave & 3;
    const int waveN = wave >> 2;
    const int blockRow = blockIdx.x * BM;
    const int blockCol = blockIdx.y * BN;

    // Staging: 16-byte chunks of bf16. 4 chunks/row, 64 rows per round.
    const int ldRow = tid >> 2;         // 0..63
    const int ldCol = (tid & 3) * 8;    // 0,8,16,24 (bf16 elements)

    f32x8 acc[2][4];
    #pragma unroll
    for (int i = 0; i < 2; ++i)
        #pragma unroll
        for (int j = 0; j < 4; ++j)
            acc[i][j] = (f32x8){};

    auto stage_tile = [&](int buf, int kt) {
        const int k0 = kt * BK;
        #pragma unroll
        for (int rr = 0; rr < 2; ++rr) {
            const int r = ldRow + rr * 64;
            const uint16_t* ga = xbf + (size_t)(blockRow + r) * D_DIM + k0 + ldCol;
            const uint16_t* gb = wbf + (size_t)(blockCol + r) * D_DIM + k0 + ldCol;
            uint16_t* la = &As[buf][r * LDK + ldCol];
            uint16_t* lb = &Bs[buf][r * LDK + ldCol];
#if defined(HAVE_ASYNC_LDS)
            __builtin_amdgcn_global_load_async_to_lds_b128(
                (g_i32x4*)ga, (l_i32x4*)la, 0, 0);
            __builtin_amdgcn_global_load_async_to_lds_b128(
                (g_i32x4*)gb, (l_i32x4*)lb, 0, 0);
#else
            *(uint4*)la = *(const uint4*)ga;
            *(uint4*)lb = *(const uint4*)gb;
#endif
        }
    };

    const int NK = D_DIM / BK;
    stage_tile(0, 0);
#if defined(HAVE_ASYNC_LDS)
    wait_async_zero();
#endif
    __syncthreads();
    for (int kt = 0; kt < NK; ++kt) {
        const int buf = kt & 1;
        if (kt + 1 < NK) stage_tile(buf ^ 1, kt + 1);
        wmma_step(As[buf], Bs[buf], lane, waveM, waveN, acc);
#if defined(HAVE_ASYNC_LDS)
        wait_async_zero();
#endif
        __syncthreads();
    }
    dist_epilogue(xs, ws, out, blockRow, blockCol, lane, waveM, waveN, acc);
}

// ---------------------------------------------------------------------------
extern "C" void kernel_launch(void* const* d_in, const int* in_sizes, int n_in,
                              void* d_out, int out_size, void* d_ws, size_t ws_size,
                              hipStream_t stream) {
    const float* x = (const float*)d_in[0];   // [B, D] fp32
    const float* W = (const float*)d_in[1];   // [C, D] fp32
    float* out = (float*)d_out;               // [B, C] fp32

    float*    xs  = (float*)d_ws;                       // [B] norms
    float*    wsn = xs + B_DIM;                         // [C] norms
    uint16_t* xbf = (uint16_t*)(wsn + C_DIM);           // [B, D] bf16
    uint16_t* wbf = xbf + (size_t)B_DIM * D_DIM;        // [C, D] bf16

    const size_t need = (size_t)(B_DIM + C_DIM) * 4
                      + ((size_t)B_DIM + C_DIM) * D_DIM * 2;

    dim3 grid(B_DIM / BM, C_DIM / BN);        // 32 x 8 workgroups

    if (ws_size >= need) {
        // Async pipeline: one fp32 pass (norms + bf16 copies), then an
        // all-bf16 WMMA GEMM whose operands stay resident in L2.
        prep_kernel<<<B_DIM + C_DIM, 256, 0, stream>>>(x, W, xs, wsn, xbf, wbf);
        dist_gemm_async_kernel<<<grid, 256, 0, stream>>>(xbf, wbf, xs, wsn, out);
    } else {
        // Fallback: convert-in-kernel GEMM, 20 KB workspace.
        norms_kernel<<<B_DIM + C_DIM, 256, 0, stream>>>(x, W, xs, wsn);
        dist_gemm_kernel<<<grid, 256, 0, stream>>>(x, W, xs, wsn, out);
    }
}
